// BaseDTA_38216619000381
// MI455X (gfx1250) — matrix operand order repaired and checked by reading, not compiled, not run on hardware
//
#include <hip/hip_runtime.h>
#include <hip/hip_bf16.h>
#include <math.h>

typedef __attribute__((ext_vector_type(16))) _Float16 v16h;
typedef __attribute__((ext_vector_type(8)))  float    v8f;

constexpr int CB = 8;      // batch
constexpr int CC = 256;    // channels
constexpr int CHD = 32;    // head dim
constexpr int CT = 1026;   // true sequence length (32*32 + lt + gt)
constexpr int CTP = 1056;  // padded to 66*16 (and 33*32)
constexpr int CN = 32;     // 4 quadrants * 8 batch sequences

// ---- WMMA fragment helpers (CDNA5 ISA 7.12.2, wave32) ----
// A 16x32 f16 frag: lane(0..15)=M; this lane's data = bytes [16h,16h+16) and
// [32+16h, +16) of its row  ->  two 128-bit loads.
__device__ __forceinline__ v16h ldfrag_a(const _Float16* rowk0, int half) {
  union { v16h v; uint4 q[2]; } u;
  const uint4* p = (const uint4*)rowk0;
  u.q[0] = p[half];
  u.q[1] = p[2 + half];
  return u.v;
}
// B 32x16 f16 frag: lane(0..15)=N; data = bytes [16h,16h+16) and [32+16h, +16) of its column row.
__device__ __forceinline__ v16h ldfrag_b(const _Float16* colk0, int half) {
  union { v16h v; uint4 q[2]; } u;
  const uint4* p = (const uint4*)colk0;
  u.q[0] = p[half];
  u.q[1] = p[2 + half];
  return u.v;
}
// element index for strided (non-vectorizable) B reads
__device__ __forceinline__ int b_kidx(int v, int p, int half) {
  return 2 * v + p + 8 * half + ((v >= 4) ? 8 : 0);
}

__device__ __forceinline__ float red_max16(float v) {
#pragma unroll
  for (int m = 1; m < 16; m <<= 1) v = fmaxf(v, __shfl_xor(v, m, 32));
  return v;
}
__device__ __forceinline__ float red_sum16(float v) {
#pragma unroll
  for (int m = 1; m < 16; m <<= 1) v += __shfl_xor(v, m, 32);
  return v;
}

// ---------------- weight f32 -> f16 ----------------
__global__ void k_wcvt(const float* w0, const float* w1, const float* w2, const float* w3,
                       _Float16* o0, _Float16* o1, _Float16* o2, _Float16* o3) {
  int i = blockIdx.x * 256 + threadIdx.x;
  int s = blockIdx.y;
  const float* src = (s == 0) ? w0 : (s == 1) ? w1 : (s == 2) ? w2 : w3;
  _Float16* dst    = (s == 0) ? o0 : (s == 1) ? o1 : (s == 2) ? o2 : o3;
  dst[i] = (_Float16)src[i];
}

// ---------------- global / local token means ----------------
__global__ void k_tokens(const float* __restrict__ x, float* __restrict__ gt,
                         float* __restrict__ lt) {
  int bc = blockIdx.x;
  int tid = threadIdx.x;
  __shared__ float sq[4][256];
  float q[4] = {0.f, 0.f, 0.f, 0.f};
  const float* xp = x + (size_t)bc * 4096;
#pragma unroll
  for (int k = 0; k < 16; ++k) {
    int pix = tid + k * 256;
    int y = pix >> 6, xc = pix & 63;
    int quad = ((y >= 32) ? 2 : 0) + ((xc >= 32) ? 1 : 0);
    q[quad] += xp[pix];
  }
#pragma unroll
  for (int qd = 0; qd < 4; ++qd) sq[qd][tid] = q[qd];
  __syncthreads();
  for (int s = 128; s > 0; s >>= 1) {
    if (tid < s) {
#pragma unroll
      for (int qd = 0; qd < 4; ++qd) sq[qd][tid] += sq[qd][tid + s];
    }
    __syncthreads();
  }
  if (tid == 0) {
    int b = bc >> 8, c = bc & 255;
    float tot = 0.f;
#pragma unroll
    for (int qd = 0; qd < 4; ++qd) {
      float s = sq[qd][0];
      tot += s;
      lt[(qd * CB + b) * CC + c] = s * (1.f / 1024.f);
    }
    gt[bc] = tot * (1.f / 4096.f);
  }
}

// ---------------- build padded f16 token sequence [CN][CTP][CC] ----------------
__global__ void k_build(const float* __restrict__ x, const float* __restrict__ gt,
                        const float* __restrict__ lt, _Float16* __restrict__ X) {
  size_t idx = (size_t)blockIdx.x * 256 + threadIdx.x;
  int c = (int)(idx % CC);
  int t = (int)((idx / CC) % CTP);
  int n = (int)(idx / ((size_t)CC * CTP));
  int q = n >> 3, b = n & 7;
  float v = 0.f;
  if (t < 1024) {
    int i = t >> 5, j = t & 31;
    int y = i + (q >> 1) * 32, xc = j + (q & 1) * 32;
    v = x[(((size_t)(b * CC + c)) * 64 + y) * 64 + xc];
  } else if (t == 1024) {
    v = lt[n * CC + c];
  } else if (t == 1025) {
    v = gt[b * CC + c];
  }
  X[idx] = (_Float16)v;
}

// ---------------- QKV projection: 8 waves share LDS-staged A tile ----------------
__global__ void k_qkv(const _Float16* __restrict__ X,
                      const _Float16* __restrict__ Wq, const _Float16* __restrict__ Wk,
                      const _Float16* __restrict__ Wv,
                      const float* bq, const float* bk, const float* bv,
                      _Float16* Q, _Float16* K, _Float16* V) {
  __shared__ __align__(16) _Float16 As[16 * 256];  // 8KB A tile
  const int tid = threadIdx.x;
  const int wv = tid >> 5, lane = tid & 31;
  const int l15 = lane & 15, half = lane >> 4;
  const int m0 = blockIdx.x * 16;
  const int sel = blockIdx.y;
  const int n = blockIdx.z;
  const _Float16* Wm = (sel == 0) ? Wq : (sel == 1) ? Wk : Wv;
  const float* bias = (sel == 0) ? bq : (sel == 1) ? bk : bv;
  _Float16* Out = (sel == 0) ? Q : (sel == 1) ? K : V;

  const _Float16* Xb = X + ((size_t)n * CTP + m0) * CC;
  {  // cooperative stage: contiguous 8KB
    const uint4* src = (const uint4*)Xb;
    uint4* dst = (uint4*)As;
    dst[tid] = src[tid];
    dst[tid + 256] = src[tid + 256];
  }
  __syncthreads();

  const int n0a = wv * 32;  // this wave owns column tiles n0a and n0a+16
  v8f acc0 = {}, acc1 = {};
#pragma unroll
  for (int ks = 0; ks < 8; ++ks) {
    v16h a = ldfrag_a(&As[l15 * CC + ks * 32], half);
    v16h b0 = ldfrag_b(&Wm[(n0a + l15) * CC + ks * 32], half);
    v16h b1 = ldfrag_b(&Wm[(n0a + 16 + l15) * CC + ks * 32], half);
    acc0 = __builtin_amdgcn_wmma_f32_16x16x32_f16(false, a, false, b0, (short)0, acc0,
                                                  false, false);
    acc1 = __builtin_amdgcn_wmma_f32_16x16x32_f16(false, a, false, b1, (short)0, acc1,
                                                  false, false);
  }
  float bc0 = bias[n0a + l15], bc1 = bias[n0a + 16 + l15];
#pragma unroll
  for (int r = 0; r < 8; ++r) {
    int trow = m0 + r + 8 * half;
    size_t ob = ((size_t)n * CTP + trow) * CC + n0a;
    Out[ob + l15] = (_Float16)(acc0[r] + bc0);
    Out[ob + 16 + l15] = (_Float16)(acc1[r] + bc1);
  }
}

// ---------------- flash MHSA: 8 q-tiles per block share K/V chunks in LDS ----------
__global__ void k_attn(const _Float16* __restrict__ Q, const _Float16* __restrict__ K,
                       const _Float16* __restrict__ V, _Float16* __restrict__ Aout) {
  __shared__ __align__(16) _Float16 Ks[32 * 32];     // 2KB keys chunk
  __shared__ __align__(16) _Float16 Vs[32 * 32];     // 2KB values chunk
  __shared__ __align__(16) _Float16 Sp[8][16 * 32];  // per-wave f16 P tiles (8KB)
  const int tid = threadIdx.x;
  const int wv = tid >> 5, lane = tid & 31;
  const int l15 = lane & 15, half = lane >> 4;
  int qt = blockIdx.x * 8 + wv;
  if (qt > CTP / 16 - 1) qt = CTP / 16 - 1;  // clamp: uniform barriers in tail block
  const int m0 = qt * 16;
  const int head = blockIdx.y;
  const int n = blockIdx.z;
  const float scale = 0.17677669529663687f;  // 1/sqrt(32)

  v16h aQ = ldfrag_a(&Q[((size_t)n * CTP + m0 + l15) * CC + head * CHD], half);

  float mst[8], lst[8];
#pragma unroll
  for (int r = 0; r < 8; ++r) { mst[r] = -1e30f; lst[r] = 0.f; }
  v8f O0 = {}, O1 = {};
  v8f zero = {};

  const int sr = tid >> 3, sp = tid & 7;  // staging: row 0..31, 4-half part 0..7
  for (int kcn = 0; kcn < 33; ++kcn) {
    const int base = kcn * 32;
    {  // cooperative stage of K,V chunk (8B per thread per buffer)
      const size_t so = ((size_t)n * CTP + base + sr) * CC + head * CHD + sp * 4;
      *(uint2*)&Ks[sr * 32 + sp * 4] = *(const uint2*)&K[so];
      *(uint2*)&Vs[sr * 32 + sp * 4] = *(const uint2*)&V[so];
    }
    if (kcn + 1 < 33)
      __builtin_prefetch(&K[((size_t)n * CTP + base + 32 + sr) * CC + head * CHD], 0, 0);
    __syncthreads();

    v16h bK0 = ldfrag_b(&Ks[l15 * 32], half);
    v16h bK1 = ldfrag_b(&Ks[(16 + l15) * 32], half);
    v8f S0 = __builtin_amdgcn_wmma_f32_16x16x32_f16(false, aQ, false, bK0, (short)0,
                                                    zero, false, false);
    v8f S1 = __builtin_amdgcn_wmma_f32_16x16x32_f16(false, aQ, false, bK1, (short)0,
                                                    zero, false, false);
    const bool mask0 = (base + l15) >= CT;
    const bool mask1 = (base + 16 + l15) >= CT;
#pragma unroll
    for (int r = 0; r < 8; ++r) {
      float s0 = mask0 ? -1e30f : S0[r] * scale;
      float s1 = mask1 ? -1e30f : S1[r] * scale;
      float rmax = red_max16(fmaxf(s0, s1));
      float mnew = fmaxf(mst[r], rmax);
      float al = __expf(mst[r] - mnew);
      float p0 = __expf(s0 - mnew), p1 = __expf(s1 - mnew);
      lst[r] = lst[r] * al + red_sum16(p0 + p1);
      mst[r] = mnew;
      O0[r] *= al;
      O1[r] *= al;
      int row = r + 8 * half;
      Sp[wv][row * 32 + l15] = (_Float16)p0;
      Sp[wv][row * 32 + 16 + l15] = (_Float16)p1;
    }
    __syncthreads();
    v16h aP = ldfrag_a(&Sp[wv][l15 * 32], half);
    v16h bV0, bV1;  // column reads of Vs: strided, element-wise
#pragma unroll
    for (int v = 0; v < 8; ++v)
#pragma unroll
      for (int p = 0; p < 2; ++p) {
        int kd = b_kidx(v, p, half);
        bV0[2 * v + p] = Vs[kd * 32 + l15];
        bV1[2 * v + p] = Vs[kd * 32 + 16 + l15];
      }
    O0 = __builtin_amdgcn_wmma_f32_16x16x32_f16(false, aP, false, bV0, (short)0, O0,
                                                false, false);
    O1 = __builtin_amdgcn_wmma_f32_16x16x32_f16(false, aP, false, bV1, (short)0, O1,
                                                false, false);
    __syncthreads();  // protect Ks/Vs before next stage
  }
#pragma unroll
  for (int r = 0; r < 8; ++r) {
    float inv = 1.f / lst[r];
    int trow = m0 + r + 8 * half;
    size_t ob = ((size_t)n * CTP + trow) * CC + head * CHD;
    Aout[ob + l15] = (_Float16)(O0[r] * inv);
    Aout[ob + 16 + l15] = (_Float16)(O1[r] * inv);
  }
}

// ---------------- output projection + residual (LDS-staged A) -> f32 Y -------------
__global__ void k_proj(const _Float16* __restrict__ A, const _Float16* __restrict__ Wo,
                       const float* bo, const _Float16* __restrict__ X,
                       float* __restrict__ Y) {
  __shared__ __align__(16) _Float16 As[16 * 256];
  const int tid = threadIdx.x;
  const int wv = tid >> 5, lane = tid & 31;
  const int l15 = lane & 15, half = lane >> 4;
  const int m0 = blockIdx.x * 16;
  const int n = blockIdx.y;
  const _Float16* Ab = A + ((size_t)n * CTP + m0) * CC;
  {
    const uint4* src = (const uint4*)Ab;
    uint4* dst = (uint4*)As;
    dst[tid] = src[tid];
    dst[tid + 256] = src[tid + 256];
  }
  __syncthreads();

  const int n0a = wv * 32;
  v8f acc0 = {}, acc1 = {};
#pragma unroll
  for (int ks = 0; ks < 8; ++ks) {
    v16h a = ldfrag_a(&As[l15 * CC + ks * 32], half);
    v16h b0 = ldfrag_b(&Wo[(n0a + l15) * CC + ks * 32], half);
    v16h b1 = ldfrag_b(&Wo[(n0a + 16 + l15) * CC + ks * 32], half);
    acc0 = __builtin_amdgcn_wmma_f32_16x16x32_f16(false, a, false, b0, (short)0, acc0,
                                                  false, false);
    acc1 = __builtin_amdgcn_wmma_f32_16x16x32_f16(false, a, false, b1, (short)0, acc1,
                                                  false, false);
  }
  float bc0 = bo[n0a + l15], bc1 = bo[n0a + 16 + l15];
#pragma unroll
  for (int r = 0; r < 8; ++r) {
    int trow = m0 + r + 8 * half;
    size_t ob = ((size_t)n * CTP + trow) * CC + n0a;
    Y[ob + l15] = acc0[r] + bc0 + (float)X[ob + l15];
    Y[ob + 16 + l15] = acc1[r] + bc1 + (float)X[ob + 16 + l15];
  }
}

// ---------------- sigmoid outer-product gate -> f_wlt [B][C][H][W] ----------------
__global__ void k_gate(const float* __restrict__ Y, const float* w_row, const float* b_row,
                       const float* w_col, const float* b_col, float* __restrict__ fwlt) {
  int idx = blockIdx.x * 256 + threadIdx.x;
  int hw = idx & 4095;
  int c = (idx >> 12) & 255;
  int b = idx >> 20;
  int y = hw >> 6, xc = hw & 63;
  int q = ((y >= 32) ? 2 : 0) + ((xc >= 32) ? 1 : 0);
  int i = y & 31, j = xc & 31;
  size_t yb = (size_t)(q * CB + b) * CTP * CC;
  float xp = Y[yb + (size_t)(i * 32 + j) * CC + c];
  float ltp = Y[yb + (size_t)1024 * CC + c];
  float rv = ltp * w_row[i] + b_row[i];
  float cv = ltp * w_col[j] + b_col[j];
  float g = 1.f / (1.f + __expf(-(rv * cv)));
  fwlt[idx] = xp * g;
}

// ---------------- fused global-token keys: kg [B][4H=256][C] f16 ----------------
__global__ void k_kglob(const float* __restrict__ Y, const float* __restrict__ Wfuse,
                        const float* bfuse, const float* w_gt, const float* b_gt,
                        _Float16* __restrict__ kg) {
  int idx = blockIdx.x * 256 + threadIdx.x;
  int o = idx & 255;
  int kidx = (idx >> 8) & 255;
  int b = idx >> 16;
  int hh = kidx >> 2, q = kidx & 3;
  const float* gtp = Y + ((size_t)(q * CB + b) * CTP + 1025) * CC;
  const float* wf = Wfuse + o * CC;
  float a1 = 0.f, a2 = 0.f;
  for (int c = 0; c < CC; ++c) {
    float w = wf[c];
    a1 += w * gtp[c];
    a2 += w;
  }
  kg[idx] = (_Float16)(w_gt[hh] * a1 + b_gt[hh] * a2 + bfuse[o]);
}

// ------- flash global attention + final add; 8 q-tiles share LDS key chunk --------
__global__ void k_gattn(const float* __restrict__ fwlt, const _Float16* __restrict__ kg,
                        float* __restrict__ out) {
  __shared__ __align__(16) _Float16 Kg[32 * 256];    // 16KB key/value chunk
  __shared__ __align__(16) _Float16 Sp[8][16 * 32];  // 8KB f16 P tiles
  const int tid = threadIdx.x;
  const int wv = tid >> 5, lane = tid & 31;
  const int l15 = lane & 15, half = lane >> 4;
  const int m0 = (blockIdx.x * 8 + wv) * 16;  // exact: 32*8 = 256 q-tiles
  const int b = blockIdx.y;
  const float scale = 0.0625f;  // 1/sqrt(256)
  v8f zero = {};

  // A fragments for all 8 k-steps (transposed f32 gather; coalesced across lanes)
  v16h aF[8];
#pragma unroll
  for (int ks = 0; ks < 8; ++ks)
#pragma unroll
    for (int e = 0; e < 16; ++e) {
      int ck = (e < 8) ? e : (8 + e);  // a_kidx pattern: 0..7 -> k, 8..15 -> k+8
      int c = ks * 32 + ck + 8 * half;
      aF[ks][e] = (_Float16)fwlt[((size_t)(b * CC + c)) * 4096 + m0 + l15];
    }

  float mst[8], lst[8];
#pragma unroll
  for (int r = 0; r < 8; ++r) { mst[r] = -1e30f; lst[r] = 0.f; }
  v8f Ov[16] = {};

  for (int kcn = 0; kcn < 8; ++kcn) {
    const int base = kcn * 32;
    {  // cooperative stage of 16KB contiguous key rows
      const uint4* src = (const uint4*)&kg[((size_t)(b * CC) + base) * CC];
      uint4* dst = (uint4*)Kg;
#pragma unroll
      for (int k = 0; k < 4; ++k) dst[tid + k * 256] = src[tid + k * 256];
    }
    __syncthreads();

    v8f S0 = zero, S1 = zero;
#pragma unroll
    for (int ks = 0; ks < 8; ++ks) {
      v16h bK0 = ldfrag_b(&Kg[l15 * CC + ks * 32], half);
      v16h bK1 = ldfrag_b(&Kg[(16 + l15) * CC + ks * 32], half);
      S0 = __builtin_amdgcn_wmma_f32_16x16x32_f16(false, aF[ks], false, bK0, (short)0,
                                                  S0, false, false);
      S1 = __builtin_amdgcn_wmma_f32_16x16x32_f16(false, aF[ks], false, bK1, (short)0,
                                                  S1, false, false);
    }
    float alr[8];
#pragma unroll
    for (int r = 0; r < 8; ++r) {
      float s0 = S0[r] * scale, s1 = S1[r] * scale;
      float rmax = red_max16(fmaxf(s0, s1));
      float mnew = fmaxf(mst[r], rmax);
      float al = __expf(mst[r] - mnew);
      float p0 = __expf(s0 - mnew), p1 = __expf(s1 - mnew);
      lst[r] = lst[r] * al + red_sum16(p0 + p1);
      mst[r] = mnew;
      alr[r] = al;
      int row = r + 8 * half;
      Sp[wv][row * 32 + l15] = (_Float16)p0;
      Sp[wv][row * 32 + 16 + l15] = (_Float16)p1;
    }
#pragma unroll
    for (int ct = 0; ct < 16; ++ct)
#pragma unroll
      for (int r = 0; r < 8; ++r) Ov[ct][r] *= alr[r];
    __syncthreads();
    v16h aP = ldfrag_a(&Sp[wv][l15 * 32], half);
#pragma unroll
    for (int ct = 0; ct < 16; ++ct) {
      v16h bV;  // column reads of Kg as V: strided, element-wise
#pragma unroll
      for (int v = 0; v < 8; ++v)
#pragma unroll
        for (int p = 0; p < 2; ++p) {
          int kd = b_kidx(v, p, half);
          bV[2 * v + p] = Kg[kd * CC + ct * 16 + l15];
        }
      Ov[ct] = __builtin_amdgcn_wmma_f32_16x16x32_f16(false, aP, false, bV, (short)0,
                                                      Ov[ct], false, false);
    }
    __syncthreads();  // protect Kg before next stage
  }
#pragma unroll
  for (int r = 0; r < 8; ++r) {
    float inv = 1.f / lst[r];
    int hw = m0 + r + 8 * half;
#pragma unroll
    for (int ct = 0; ct < 16; ++ct) {
      int c = ct * 16 + l15;
      size_t oi = ((size_t)(b * CC + c)) * 4096 + hw;
      out[oi] = Ov[ct][r] * inv + fwlt[oi];
    }
  }
}

extern "C" void kernel_launch(void* const* d_in, const int* in_sizes, int n_in,
                              void* d_out, int out_size, void* d_ws, size_t ws_size,
                              hipStream_t stream) {
  (void)in_sizes; (void)n_in; (void)out_size; (void)ws_size;
  const float* x     = (const float*)d_in[0];
  const float* Wq    = (const float*)d_in[1];
  const float* bq    = (const float*)d_in[2];
  const float* Wk    = (const float*)d_in[3];
  const float* bk    = (const float*)d_in[4];
  const float* Wv    = (const float*)d_in[5];
  const float* bv    = (const float*)d_in[6];
  const float* Wo    = (const float*)d_in[7];
  const float* bo    = (const float*)d_in[8];
  const float* w_row = (const float*)d_in[9];
  const float* b_row = (const float*)d_in[10];
  const float* w_col = (const float*)d_in[11];
  const float* b_col = (const float*)d_in[12];
  const float* w_gt  = (const float*)d_in[13];
  const float* b_gt  = (const float*)d_in[14];
  const float* Wfuse = (const float*)d_in[15];
  const float* bfuse = (const float*)d_in[16];

  char* ws = (char*)d_ws;
  size_t off = 0;
  auto alloc = [&](size_t bytes) {
    size_t o = off;
    off += (bytes + 255) & ~(size_t)255;
    return o;
  };
  const size_t seqE = (size_t)CN * CTP * CC;
  _Float16* Wq16 = (_Float16*)(ws + alloc(65536 * 2));
  _Float16* Wk16 = (_Float16*)(ws + alloc(65536 * 2));
  _Float16* Wv16 = (_Float16*)(ws + alloc(65536 * 2));
  _Float16* Wo16 = (_Float16*)(ws + alloc(65536 * 2));
  float*    gt   = (float*)(ws + alloc((size_t)CB * CC * 4));
  float*    lt   = (float*)(ws + alloc((size_t)4 * CB * CC * 4));
  _Float16* Xs   = (_Float16*)(ws + alloc(seqE * 2));
  _Float16* Qb   = (_Float16*)(ws + alloc(seqE * 2));
  _Float16* Kb   = (_Float16*)(ws + alloc(seqE * 2));
  _Float16* Vb   = (_Float16*)(ws + alloc(seqE * 2));
  _Float16* Ab   = (_Float16*)(ws + alloc(seqE * 2));
  float*    Y    = (float*)(ws + alloc(seqE * 4));
  float*    fwlt = (float*)(ws + alloc((size_t)CB * CC * 4096 * 4));
  _Float16* kgb  = (_Float16*)(ws + alloc((size_t)CB * 256 * CC * 2));

  k_wcvt<<<dim3(256, 4), 256, 0, stream>>>(Wq, Wk, Wv, Wo, Wq16, Wk16, Wv16, Wo16);
  k_tokens<<<CB * CC, 256, 0, stream>>>(x, gt, lt);
  k_build<<<(unsigned)(seqE / 256), 256, 0, stream>>>(x, gt, lt, Xs);
  k_qkv<<<dim3(CTP / 16, 3, CN), 256, 0, stream>>>(Xs, Wq16, Wk16, Wv16, bq, bk, bv,
                                                   Qb, Kb, Vb);
  k_attn<<<dim3((CTP / 16 + 7) / 8, 8, CN), 256, 0, stream>>>(Qb, Kb, Vb, Ab);
  k_proj<<<dim3(CTP / 16, CN), 256, 0, stream>>>(Ab, Wo16, bo, Xs, Y);
  k_gate<<<CB * CC * 4096 / 256, 256, 0, stream>>>(Y, w_row, b_row, w_col, b_col, fwlt);
  k_kglob<<<CB * 256 * CC / 256, 256, 0, stream>>>(Y, Wfuse, bfuse, w_gt, b_gt, kgb);
  k_gattn<<<dim3(32, CB), 256, 0, stream>>>(fwlt, kgb, (float*)d_out);
}
